// BahdanauAttention_3607772529398
// MI455X (gfx1250) — matrix operand order repaired and checked by reading, not compiled, not run on hardware
//
#include <hip/hip_runtime.h>
#include <hip/hip_bf16.h>

typedef __attribute__((ext_vector_type(16))) __bf16 v16bf;
typedef __attribute__((ext_vector_type(8)))  float  v8f;
typedef __attribute__((ext_vector_type(4)))  int    v4i;

union ABv { uint4 q[2]; v16bf v; };

#if __has_builtin(__builtin_amdgcn_global_load_async_to_lds_b128)
#define HAVE_ASYNC_LDS 1
#else
#define HAVE_ASYNC_LDS 0
#endif

#if HAVE_ASYNC_LDS
typedef __attribute__((address_space(1))) v4i* gptr_b128;  // global int4*
typedef __attribute__((address_space(3))) v4i* lptr_b128;  // LDS int4*
#endif

__device__ __forceinline__ unsigned pk_bf16(float a, float b) {
#if __has_builtin(__builtin_amdgcn_cvt_pk_bf16_f32)
    auto r = __builtin_amdgcn_cvt_pk_bf16_f32(a, b);
    return __builtin_bit_cast(unsigned, r);
#else
    unsigned short x = __builtin_bit_cast(unsigned short, (__bf16)a);
    unsigned short y = __builtin_bit_cast(unsigned short, (__bf16)b);
    return (unsigned)x | ((unsigned)y << 16);
#endif
}

// async 16B copy global -> LDS (ASYNCcnt path); fallback: sync copy via VGPRs
__device__ __forceinline__ void async_b128(unsigned* lds_p, const unsigned* g_p) {
#if HAVE_ASYNC_LDS
    gptr_b128 gp = (gptr_b128)(unsigned long long)(uintptr_t)g_p;
    lptr_b128 lp = (lptr_b128)(unsigned)(uintptr_t)lds_p; // flat LDS addr[31:0] = LDS offset
    __builtin_amdgcn_global_load_async_to_lds_b128(gp, lp, 0, 0);
#else
    *(uint4*)lds_p = *(const uint4*)g_p;
#endif
}

__device__ __forceinline__ void wait_async0() {
#if HAVE_ASYNC_LDS
#if __has_builtin(__builtin_amdgcn_s_wait_asynccnt)
    __builtin_amdgcn_s_wait_asynccnt(0);
#else
    asm volatile("s_wait_asynccnt 0x0" ::: "memory");
#endif
#endif
}

// ---------------------------------------------------------------------------
// K1: hb[b,h] = sum_k query[b,k]*W1[h,k] + W1_b[h] + W2_b[h]
// ---------------------------------------------------------------------------
__global__ __launch_bounds__(256)
void hidden_kernel(const float* __restrict__ q, const float* __restrict__ W1,
                   const float* __restrict__ b1, const float* __restrict__ b2,
                   float* __restrict__ hb) {
    __shared__ float ldsQ[1024];
    const int b = blockIdx.x, tid = threadIdx.x;
    for (int i = tid; i < 1024; i += 256) ldsQ[i] = q[b * 1024 + i];
    __syncthreads();
    for (int j = 0; j < 4; ++j) {
        const int h = tid + j * 256;
        const float* wr = W1 + (size_t)h * 1024;
        float acc = 0.f;
        for (int k = 0; k < 1024; k += 4) {
            float4 w4 = *(const float4*)(wr + k);
            acc += w4.x * ldsQ[k] + w4.y * ldsQ[k + 1] +
                   w4.z * ldsQ[k + 2] + w4.w * ldsQ[k + 3];
        }
        hb[b * 1024 + h] = acc + b1[h] + b2[h];
    }
}

// ---------------------------------------------------------------------------
// K2: w2t[k][h] = bf16(W2[h][k])   (transpose + convert, LDS tiled)
// ---------------------------------------------------------------------------
__global__ __launch_bounds__(256)
void w2cvt_kernel(const float* __restrict__ W2, __bf16* __restrict__ w2t) {
    __shared__ float tile[32][33];
    const int kb = blockIdx.x * 32;
    const int ht = blockIdx.y * 32;
    const int tx = threadIdx.x & 31, ty = threadIdx.x >> 5;  // 32x8
    for (int i = 0; i < 32; i += 8)
        tile[ty + i][tx] = W2[(size_t)(ht + ty + i) * 1280 + kb + tx];
    __syncthreads();
    for (int i = 0; i < 32; i += 8)
        w2t[(size_t)(kb + ty + i) * 1024 + ht + tx] = (__bf16)tile[tx][ty + i];
}

// ---------------------------------------------------------------------------
// K3: fused  scores[b,l] = Va . relu(hb[b,:] + keys[b,l,:] @ W2^T) + Va_b
// 256 thr = 8 waves = 2 M-groups (32 rows) x 4 N-groups (64 cols).
// 160-step software pipeline: double-buffered LDS, async B staging,
// one barrier per K-step, 8 x v_wmma_f32_16x16x32_bf16 per step per wave.
// ---------------------------------------------------------------------------
__global__ __launch_bounds__(256, 2)
void attn_scores_kernel(const float* __restrict__ keys,
                        const float* __restrict__ hbias,
                        const __bf16* __restrict__ w2t,
                        const float* __restrict__ va,
                        const float* __restrict__ vab,
                        float* __restrict__ scores) {
    __shared__ float    scoreLds[64];
    __shared__ unsigned ldsA[2][64 * 20];    // 64 rows x 16 dw (K=32 bf16), stride 20
    __shared__ unsigned ldsB[2][32 * 132];   // 32 K-rows x 128 dw (N=256 bf16), stride 132

    const int b    = blockIdx.y;
    const int l0   = blockIdx.x * 64;
    const int tid  = threadIdx.x;
    const int wave = tid >> 5, lane = tid & 31;
    const int hlf  = lane >> 4, col = lane & 15;
    const int ngrp = wave & 3, mgrp = wave >> 2;
    const int mbase = mgrp * 32;
    const int nw    = ngrp * 64;

    if (tid < 64) scoreLds[tid] = 0.f;

    const float*    keyrow = keys + ((size_t)b * 2048 + l0) * 1280;
    const unsigned* w2u    = (const unsigned*)w2t;   // [1280][512] dwords

    const int arow = tid >> 2, adw = (tid & 3) * 4;   // A: 64 rows x 4 dw-groups
    const int brow = tid >> 3, bdw = (tid & 7) * 16;  // B: 32 rows x 16 dw each

    // stage step s into buffer (s&1):  p = s/40 (H-pass), k0 = (s%40)*32
    auto stage = [&](int s) {
        const int buf = s & 1;
        const int p   = s / 40;
        const int k0  = (s - p * 40) * 32;
        {   // A: keys f32 -> bf16 pairs (needs VALU cvt, manual path)
            const float* src = keyrow + (size_t)arow * 1280 + k0 + adw * 2;
            float4 f0 = *(const float4*)(src);
            float4 f1 = *(const float4*)(src + 4);
            uint4 qd = make_uint4(pk_bf16(f0.x, f0.y), pk_bf16(f0.z, f0.w),
                                  pk_bf16(f1.x, f1.y), pk_bf16(f1.z, f1.w));
            *(uint4*)&ldsA[buf][arow * 20 + adw] = qd;
        }
        {   // B: pure byte copy -> async global->LDS (ASYNCcnt)
            const unsigned* src = w2u + (size_t)(k0 + brow) * 512 + p * 128 + bdw;
            unsigned* dst = &ldsB[buf][brow * 132 + bdw];
            async_b128(dst,      src);
            async_b128(dst + 4,  src + 4);
            async_b128(dst + 8,  src + 8);
            async_b128(dst + 12, src + 12);
        }
    };

    v8f acc[2][4] = {};
    stage(0);

    for (int s = 0; s < 160; ++s) {
        const int buf = s & 1;
        wait_async0();        // our async copies into `buf` have landed
        __syncthreads();      // everyone's copies landed; prev reads of buf^1 done
        if (s + 1 < 160) stage(s + 1);   // fill buf^1 while we compute on buf

        // A regs per ISA 16-bit A layout: lane m row, halves split K 8/16
        ABv aR[2];
#pragma unroll
        for (int mt = 0; mt < 2; ++mt) {
            const int base = (mbase + mt * 16 + col) * 20 + hlf * 4;
            aR[mt].q[0] = *(const uint4*)&ldsA[buf][base];
            aR[mt].q[1] = *(const uint4*)&ldsA[buf][base + 8];
        }
        // B regs: N = lane%16, elements K in table order (halves split K 8/16)
        ABv bR[4];
        const unsigned short* lbh = (const unsigned short*)&ldsB[buf][0];
#pragma unroll
        for (int nt = 0; nt < 4; ++nt) {
            const int n = nw + nt * 16 + col;
#pragma unroll
            for (int s2 = 0; s2 < 16; ++s2) {
                const int k = (s2 < 8) ? (8 * hlf + s2) : (16 + 8 * hlf + (s2 - 8));
                bR[nt].v[s2] = __builtin_bit_cast(__bf16, lbh[k * 264 + n]);
            }
        }
#pragma unroll
        for (int nt = 0; nt < 4; ++nt)
#pragma unroll
            for (int mt = 0; mt < 2; ++mt)
                acc[mt][nt] = __builtin_amdgcn_wmma_f32_16x16x32_bf16(
                    false, aR[mt].v, false, bR[nt].v,
                    (short)0, acc[mt][nt], false, false);

        if ((s % 40) == 39) {
            // end of H-pass p: relu(C + hb) * Va, reduce 64 cols of this pass
            const int p = s / 40;
            const int ncol0 = p * 256;
            float hbv[4], vav[4];
#pragma unroll
            for (int nt = 0; nt < 4; ++nt) {
                const int h = ncol0 + nw + nt * 16 + col;
                hbv[nt] = hbias[b * 1024 + h];
                vav[nt] = va[h];
            }
#pragma unroll
            for (int mt = 0; mt < 2; ++mt) {
#pragma unroll
                for (int r = 0; r < 8; ++r) {
                    float t = 0.f;
#pragma unroll
                    for (int nt = 0; nt < 4; ++nt) {
                        float v = acc[mt][nt][r] + hbv[nt];
                        v = v > 0.f ? v : 0.f;
                        t += v * vav[nt];
                    }
                    t += __shfl_xor(t, 1, 32);
                    t += __shfl_xor(t, 2, 32);
                    t += __shfl_xor(t, 4, 32);
                    t += __shfl_xor(t, 8, 32);
                    if (col == 0) {
                        const int row = mbase + mt * 16 + 8 * hlf + r;
                        atomicAdd(&scoreLds[row], t);
                    }
                }
            }
#pragma unroll
            for (int mt = 0; mt < 2; ++mt)
#pragma unroll
                for (int nt = 0; nt < 4; ++nt)
                    acc[mt][nt] = (v8f){};
        }
    }
    __syncthreads();
    if (tid < 64)
        scores[(size_t)b * 2048 + l0 + tid] = scoreLds[tid] + vab[0];
}

// ---------------------------------------------------------------------------
// K4: softmax over L=2048 per batch row
// ---------------------------------------------------------------------------
__global__ __launch_bounds__(256)
void softmax_kernel(const float* __restrict__ scores, float* __restrict__ weights) {
    __shared__ float red[256];
    const int b = blockIdx.x, tid = threadIdx.x;
    const float* s = scores + (size_t)b * 2048;
    float vals[8], m = -1e30f;
    for (int j = 0; j < 8; ++j) { vals[j] = s[tid + j * 256]; m = fmaxf(m, vals[j]); }
    red[tid] = m; __syncthreads();
    for (int o = 128; o > 0; o >>= 1) {
        if (tid < o) red[tid] = fmaxf(red[tid], red[tid + o]);
        __syncthreads();
    }
    m = red[0]; __syncthreads();
    float sum = 0.f;
    for (int j = 0; j < 8; ++j) { vals[j] = __expf(vals[j] - m); sum += vals[j]; }
    red[tid] = sum; __syncthreads();
    for (int o = 128; o > 0; o >>= 1) {
        if (tid < o) red[tid] += red[tid + o];
        __syncthreads();
    }
    const float inv = 1.0f / red[0];
    for (int j = 0; j < 8; ++j)
        weights[(size_t)b * 2048 + tid + j * 256] = vals[j] * inv;
}

// ---------------------------------------------------------------------------
// K5: context partials over L-chunks; grid (5 kchunks, 8 lchunks, 32 b)
// ---------------------------------------------------------------------------
__global__ __launch_bounds__(256)
void ctx_partial_kernel(const float* __restrict__ keys,
                        const float* __restrict__ weights,
                        float* __restrict__ partial) {
    __shared__ float wl[256];
    const int kc = blockIdx.x, lc = blockIdx.y, b = blockIdx.z;
    const int tid = threadIdx.x;
    wl[tid] = weights[(size_t)b * 2048 + lc * 256 + tid];
    __syncthreads();
    const int k = kc * 256 + tid;
    const float* kp = keys + ((size_t)b * 2048 + lc * 256) * 1280 + k;
    float acc = 0.f;
#pragma unroll 4
    for (int l = 0; l < 256; ++l)
        acc += kp[(size_t)l * 1280] * wl[l];
    partial[((size_t)lc * 32 + b) * 1280 + k] = acc;
}

// K6: reduce 8 partials -> context
__global__ __launch_bounds__(256)
void ctx_reduce_kernel(const float* __restrict__ partial, float* __restrict__ context) {
    const int idx = blockIdx.x * 256 + threadIdx.x;  // 0..40959
    float acc = 0.f;
    for (int lc = 0; lc < 8; ++lc) acc += partial[(size_t)lc * 40960 + idx];
    context[idx] = acc;
}

// ---------------------------------------------------------------------------
extern "C" void kernel_launch(void* const* d_in, const int* in_sizes, int n_in,
                              void* d_out, int out_size, void* d_ws, size_t ws_size,
                              hipStream_t stream) {
    const float* query = (const float*)d_in[0];
    const float* keys  = (const float*)d_in[1];
    const float* W1w   = (const float*)d_in[2];
    const float* W1b   = (const float*)d_in[3];
    const float* W2w   = (const float*)d_in[4];
    const float* W2b   = (const float*)d_in[5];
    const float* Vaw   = (const float*)d_in[6];
    const float* Vab   = (const float*)d_in[7];

    float* out      = (float*)d_out;
    float* context  = out;               // [32,1280]
    float* weights  = out + 32 * 1280;   // [32,2048]

    float*  hb      = (float*)d_ws;                        // 32768 f
    float*  scores  = hb + 32 * 1024;                      // 65536 f
    float*  partial = scores + 32 * 2048;                  // 8*32*1280 f
    __bf16* w2t     = (__bf16*)(partial + 8 * 32 * 1280);  // 1280*1024 bf16

    hidden_kernel<<<32, 256, 0, stream>>>(query, W1w, W1b, W2b, hb);
    w2cvt_kernel<<<dim3(40, 32), 256, 0, stream>>>(W2w, w2t);
    attn_scores_kernel<<<dim3(32, 32), 256, 0, stream>>>(keys, hb, w2t, Vaw, Vab, scores);
    softmax_kernel<<<32, 256, 0, stream>>>(scores, weights);
    ctx_partial_kernel<<<dim3(5, 8, 32), 256, 0, stream>>>(keys, weights, partial);
    ctx_reduce_kernel<<<160, 256, 0, stream>>>(partial, context);
}
